// MoEHead_39067022524652
// MI455X (gfx1250) — matrix operand order repaired and checked by reading, not compiled, not run on hardware
//
#include <hip/hip_runtime.h>
#include <math.h>

#define B_N   16384
#define D_IN  1024
#define HID   2048
#define NC    8
#define DZ    256

#define MT        128
#define NCHUNK    128
#define XS_STRIDE 1032   // 1024 + 8 bf16 pad -> conflict-free b128 DS reads
#define HS_STRIDE 136    // 128 + 8 bf16 pad -> conflict-free b128 DS reads

typedef __attribute__((ext_vector_type(16))) __bf16 v16bf;
typedef __attribute__((ext_vector_type(8)))  float  v8f;
typedef __attribute__((ext_vector_type(4)))  __bf16 v4bf;

union FragBF { v16bf v; uint4 q[2]; };

__device__ __forceinline__ float wred(float v) {
#pragma unroll
  for (int m = 16; m; m >>= 1) v += __shfl_xor(v, m, 32);
  return v;
}

// ---------------------------------------------------------------------------
// Gate: cosine-sim softmax over 8 experts; also seeds logits with gated b2.
// One wave per row.
// ---------------------------------------------------------------------------
__global__ void __launch_bounds__(256) gate_kernel(
    const float* __restrict__ z, const float* __restrict__ mu,
    const float* __restrict__ b2, const int* __restrict__ tau_p,
    float* __restrict__ logits, float* __restrict__ wout) {
  int w = threadIdx.x >> 5, lane = threadIdx.x & 31;
  int row = blockIdx.x * 8 + w;
  const float* zr = z + (size_t)row * DZ;
  float zv[8]; float zsq = 0.f;
#pragma unroll
  for (int i = 0; i < 8; ++i) { zv[i] = zr[lane * 8 + i]; zsq += zv[i] * zv[i]; }
  zsq = wred(zsq);
  float nz = fmaxf(sqrtf(zsq), 1e-12f);
  float s[8];
#pragma unroll
  for (int c = 0; c < 8; ++c) {
    const float* mr = mu + c * DZ + lane * 8;
    float d = 0.f, m2 = 0.f;
#pragma unroll
    for (int i = 0; i < 8; ++i) { float mv = mr[i]; d += zv[i] * mv; m2 += mv * mv; }
    d = wred(d); m2 = wred(m2);
    s[c] = d / (nz * fmaxf(sqrtf(m2), 1e-12f));
  }
  float tau = fmaxf(1e-6f, (float)tau_p[0]);
  float mx = -1e30f;
#pragma unroll
  for (int c = 0; c < 8; ++c) { s[c] /= tau; mx = fmaxf(mx, s[c]); }
  float sum = 0.f;
#pragma unroll
  for (int c = 0; c < 8; ++c) { s[c] = __expf(s[c] - mx); sum += s[c]; }
  float inv = 1.f / sum;
#pragma unroll
  for (int c = 0; c < 8; ++c) s[c] *= inv;
  if (lane < 8) {
    float wv = s[0];
#pragma unroll
    for (int c = 1; c < 8; ++c) if (lane == c) wv = s[c];
    wout[(size_t)row * NC + lane] = wv;
    float lb = 0.f;
#pragma unroll
    for (int e = 0; e < 8; ++e) lb += s[e] * b2[e * NC + lane];
    logits[(size_t)row * NC + lane] = lb;
  }
}

// ---------------------------------------------------------------------------
// LayerNorm (shared across experts) -> bf16 xhat. One block per row.
// ---------------------------------------------------------------------------
__global__ void __launch_bounds__(256) ln_kernel(const float* __restrict__ feat,
                                                 __bf16* __restrict__ xhat) {
  int row = blockIdx.x, tid = threadIdx.x;
  const float4 v = *(const float4*)(feat + (size_t)row * D_IN + tid * 4);
  float sum = v.x + v.y + v.z + v.w;
  float sq  = v.x * v.x + v.y * v.y + v.z * v.z + v.w * v.w;
  sum = wred(sum); sq = wred(sq);
  __shared__ float reds[8], redq[8];
  int w = tid >> 5, lane = tid & 31;
  if (lane == 0) { reds[w] = sum; redq[w] = sq; }
  __syncthreads();
  float ts = 0.f, tq = 0.f;
#pragma unroll
  for (int i = 0; i < 8; ++i) { ts += reds[i]; tq += redq[i]; }
  float mean = ts * (1.f / D_IN);
  float var  = tq * (1.f / D_IN) - mean * mean;
  float rstd = rsqrtf(var + 1e-5f);
  v4bf o;
  o[0] = (__bf16)((v.x - mean) * rstd);
  o[1] = (__bf16)((v.y - mean) * rstd);
  o[2] = (__bf16)((v.z - mean) * rstd);
  o[3] = (__bf16)((v.w - mean) * rstd);
  *(v4bf*)(xhat + (size_t)row * D_IN + tid * 4) = o;
}

// ---------------------------------------------------------------------------
// W1t[e][h][d] = bf16(gamma[e][d] * W1[e][d][h])  (tiled transpose via LDS)
// ---------------------------------------------------------------------------
__global__ void __launch_bounds__(256) w1t_kernel(const float* __restrict__ W1,
                                                  const float* __restrict__ gamma,
                                                  __bf16* __restrict__ W1t) {
  __shared__ float tile[32][33];
  int h0 = blockIdx.x * 32, d0 = blockIdx.y * 32, e = blockIdx.z;
  int tx = threadIdx.x, ty = threadIdx.y;  // (32, 8)
#pragma unroll
  for (int ii = 0; ii < 4; ++ii) {
    int d = d0 + ty + 8 * ii;
    tile[ty + 8 * ii][tx] =
        gamma[e * D_IN + d] * W1[((size_t)e * D_IN + d) * HID + h0 + tx];
  }
  __syncthreads();
#pragma unroll
  for (int ii = 0; ii < 4; ++ii) {
    int h = h0 + ty + 8 * ii;
    W1t[((size_t)e * HID + h) * D_IN + d0 + tx] = (__bf16)tile[tx][ty + 8 * ii];
  }
}

// ---------------------------------------------------------------------------
// b1p[e][h] = b1[e][h] + sum_d beta[e][d] * W1[e][d][h]
// ---------------------------------------------------------------------------
__global__ void __launch_bounds__(256) b1p_kernel(const float* __restrict__ W1,
                                                  const float* __restrict__ beta,
                                                  const float* __restrict__ b1,
                                                  float* __restrict__ b1p) {
  int e = blockIdx.y;
  int h = blockIdx.x * 256 + threadIdx.x;
  float s = b1[e * HID + h];
  const float* wp = W1 + (size_t)e * D_IN * HID + h;
  const float* bp = beta + e * D_IN;
  for (int d = 0; d < D_IN; ++d) s += bp[d] * wp[(size_t)d * HID];
  b1p[e * HID + h] = s;
}

// ---------------------------------------------------------------------------
// Fused expert kernel.
// grid = (B/128 tiles, 8 experts); 256 threads = 8 waves.
// Per 128-wide H-chunk:
//   GEMM1: each wave owns a DISJOINT 16-wide N column x full M=128
//          (8 c-tiles) -> zero cross-wave redundancy on W1t global loads;
//          A-frags from LDS X tile via FOUR laundered base pointers so every
//          ds_load_b128 uses a 16-bit immediate offset (no in-loop v_adds).
//   bias+ReLU -> bf16 Hs -> GEMM2 (K=128) into persistent 16x16 acc.
// Epilogue: gate-scaled atomicAdd into logits.
// ---------------------------------------------------------------------------
__global__ void __launch_bounds__(256) moe_main_kernel(
    const __bf16* __restrict__ xhat, const __bf16* __restrict__ W1t,
    const float* __restrict__ b1p, const float* __restrict__ W2g,
    const float* __restrict__ gate, float* __restrict__ logits) {
  __shared__ __bf16 Xs[MT * XS_STRIDE];      // 258 KB
  __shared__ __bf16 Hs[MT * HS_STRIDE];      // 34 KB
  __shared__ __bf16 W2s[16 * HS_STRIDE];     // 4.25 KB (transposed W2 chunk)

  int tid  = threadIdx.x;
  int lane = tid & 31;
  int w    = tid >> 5;               // wave id 0..7 = owned N column / M row16
  int l15  = lane & 15, lhi = lane >> 4;
  int e    = blockIdx.y;
  int rowBase = blockIdx.x * MT;

  // ---- load X tile (bf16) into LDS once ----
  const __bf16* xg = xhat + (size_t)rowBase * D_IN;
  for (int i = tid; i < MT * (D_IN / 8); i += 256) {
    int r = i >> 7;
    int c = (i & 127) * 8;
    *(uint4*)(&Xs[r * XS_STRIDE + c]) = *(const uint4*)(xg + (size_t)r * D_IN + c);
  }
  __syncthreads();

  const v8f z8 = {0.f, 0.f, 0.f, 0.f, 0.f, 0.f, 0.f, 0.f};
  v8f acc = z8;  // GEMM2 accumulator: wave w owns output rows [16w, 16w+16)

  const int aoff = lhi * 8;    // A frag: lane-lo K{0..7,16..23}, lane-hi K{8..15,24..31}
  const int boff = lhi * 8;    // B frag: lane-lo K{0..7,16..23}, lane-hi K{8..15,24..31}

  // Four A base pointers, each covering TWO 16-row M tiles: folded DS offsets
  // stay <= ~51KB (16-bit immediate). The empty asm makes each offset an
  // opaque VGPR so LLVM cannot re-associate them into one base + huge offset.
  const __bf16* Ap[4];
#pragma unroll
  for (int i = 0; i < 4; ++i) {
    int moff = i * (2 * 16 * XS_STRIDE);
    asm("" : "+v"(moff));
    Ap[i] = &Xs[l15 * XS_STRIDE] + moff;
  }

  const __bf16* HA = &Hs[(16 * w + l15) * HS_STRIDE];
  const __bf16* WB = &W2s[l15 * HS_STRIDE];

  for (int ch = 0; ch < HID; ch += NCHUNK) {
    v8f cc[8];
#pragma unroll
    for (int m = 0; m < 8; ++m) cc[m] = z8;

    // wave-exclusive B column group: h = ch + 16*w + l15
    const __bf16* Bb = W1t + ((size_t)e * HID + ch + 16 * w + l15) * D_IN;

    // ---- GEMM1: 128(M) x 16(N, per wave) x 1024(K) ----
#pragma unroll 4
    for (int kb = 0; kb < D_IN; kb += 32) {
      FragBF b;
      b.q[0] = *(const uint4*)(Bb + kb + boff);
      b.q[1] = *(const uint4*)(Bb + kb + boff + 16);
#pragma unroll
      for (int m = 0; m < 8; ++m) {
        const __bf16* A_m = Ap[m >> 1] + (m & 1) * (16 * XS_STRIDE);
        FragBF a;
        a.q[0] = *(const uint4*)(A_m + kb + aoff);
        a.q[1] = *(const uint4*)(A_m + kb + aoff + 16);
        cc[m] = __builtin_amdgcn_wmma_f32_16x16x32_bf16(false, a.v, false, b.v,
                                                        (short)0, cc[m], false, false);
      }
    }

    float bj = b1p[e * HID + ch + 16 * w + l15];  // one bias col per lane

    __syncthreads();  // previous chunk's GEMM2 finished reading Hs/W2s

    // ---- bias + ReLU + bf16 -> Hs (wave writes its exclusive 16 cols) ----
    int col = 16 * w + l15;
#pragma unroll
    for (int m = 0; m < 8; ++m) {
#pragma unroll
      for (int r = 0; r < 8; ++r) {
        int row = 16 * m + r + 8 * lhi;
        Hs[row * HS_STRIDE + col] = (__bf16)fmaxf(cc[m][r] + bj, 0.f);
      }
    }
    // ---- W2 chunk, transposed (only rows c<8 valid; pad cols discarded) ----
#pragma unroll
    for (int it = 0; it < 4; ++it) {
      int idx = tid + it * 256;
      int k = idx >> 3, c = idx & 7;
      W2s[c * HS_STRIDE + k] = (__bf16)W2g[((size_t)e * HID + ch + k) * NC + c];
    }
    __syncthreads();

    // ---- GEMM2: acc[16x16] += H[16x128] @ W2[128x16] ----
#pragma unroll
    for (int kb = 0; kb < NCHUNK; kb += 32) {
      FragBF a, b;
      a.q[0] = *(const uint4*)(HA + kb + aoff);
      a.q[1] = *(const uint4*)(HA + kb + aoff + 16);
      b.q[0] = *(const uint4*)(WB + kb + boff);
      b.q[1] = *(const uint4*)(WB + kb + boff + 16);
      acc = __builtin_amdgcn_wmma_f32_16x16x32_bf16(false, a.v, false, b.v,
                                                    (short)0, acc, false, false);
    }
  }

  // ---- gate-scale + accumulate into logits ----
  if (l15 < 8) {
#pragma unroll
    for (int r = 0; r < 8; ++r) {
      int row = rowBase + 16 * w + r + 8 * lhi;
      float gv = gate[(size_t)row * NC + e];
      atomicAdd(&logits[(size_t)row * NC + l15], acc[r] * gv);
    }
  }
}

// ---------------------------------------------------------------------------
extern "C" void kernel_launch(void* const* d_in, const int* in_sizes, int n_in,
                              void* d_out, int out_size, void* d_ws, size_t ws_size,
                              hipStream_t stream) {
  const float* feat  = (const float*)d_in[0];
  const float* z     = (const float*)d_in[1];
  const float* mu    = (const float*)d_in[2];
  const float* gamma = (const float*)d_in[3];
  const float* beta  = (const float*)d_in[4];
  const float* W1    = (const float*)d_in[5];
  const float* b1    = (const float*)d_in[6];
  const float* W2    = (const float*)d_in[7];
  const float* b2    = (const float*)d_in[8];
  const int*   tau   = (const int*)d_in[9];

  float* logits = (float*)d_out;                       // [B, 8]
  float* wout   = logits + (size_t)B_N * NC;           // [B, 8]

  char* ws = (char*)d_ws;
  __bf16* xhat = (__bf16*)ws;                                          // 32 MB
  __bf16* W1t  = (__bf16*)(ws + (size_t)B_N * D_IN * 2);               // 32 MB
  float*  b1p  = (float*)(ws + (size_t)B_N * D_IN * 2 +
                               (size_t)NC * HID * D_IN * 2);           // 64 KB

  gate_kernel<<<B_N / 8, 256, 0, stream>>>(z, mu, b2, tau, logits, wout);
  ln_kernel<<<B_N, 256, 0, stream>>>(feat, xhat);
  w1t_kernel<<<dim3(HID / 32, D_IN / 32, NC), dim3(32, 8), 0, stream>>>(W1, gamma, W1t);
  b1p_kernel<<<dim3(HID / 256, NC), 256, 0, stream>>>(W1, beta, b1, b1p);
  moe_main_kernel<<<dim3(B_N / MT, NC), 256, 0, stream>>>(xhat, W1t, b1p, W2, wout, logits);
}
